// GraphormerLayer_9491877724299
// MI455X (gfx1250) — matrix operand order repaired and checked by reading, not compiled, not run on hardware
//
#include <hip/hip_runtime.h>
#include <hip/hip_bf16.h>

// ---------------------------------------------------------------------------
// Graphormer layer for MI455X (gfx1250, wave32, WMMA + async LDS fills).
// GEMMs: v_wmma_f32_16x16x32_f16, double-buffered LDS tiles filled with
// global_load_async_to_lds_b128 (ASYNCcnt-tracked, overlapped with compute).
// ---------------------------------------------------------------------------

typedef _Float16 v16h __attribute__((ext_vector_type(16)));
typedef _Float16 h8   __attribute__((ext_vector_type(8)));
typedef float    v8f  __attribute__((ext_vector_type(8)));

#define HIDDEN   512
#define HEADS    8
#define HEAD_DIM 64
#define NB       16          // batch
#define NSEQ     512         // sequence length
#define MROWS    (NB * NSEQ) // 8192
#define FFN_DIM  2048

// Issue one 16-byte global->LDS async copy (per lane). dsaddr is the LDS byte
// offset (low 32 bits of a generic pointer into the group segment).
__device__ __forceinline__ void async_copy_b128(unsigned lds_off,
                                                const void* gptr) {
  unsigned long long ga = (unsigned long long)(uintptr_t)gptr;
  asm volatile("global_load_async_to_lds_b128 %0, %1, off"
               :: "v"(lds_off), "v"(ga)
               : "memory");
}
__device__ __forceinline__ void wait_async_zero() {
  asm volatile("s_wait_asynccnt 0x0" ::: "memory");
}

// ---------------------------------------------------------------------------
// Elementwise f32 -> f16 conversion
// ---------------------------------------------------------------------------
__global__ void cvt_f32_f16(const float* __restrict__ src,
                            _Float16* __restrict__ dst, int n) {
  int i = blockIdx.x * blockDim.x + threadIdx.x;
  int stride = gridDim.x * blockDim.x;
  for (; i < n; i += stride) dst[i] = (_Float16)src[i];
}

// ---------------------------------------------------------------------------
// Tiled WMMA GEMM:  out[M,N] = A[M,K] (f16) @ W[K,N] (f16) + bias
// Block tile 128x128, 8 waves (wave: 16 rows x 128 cols = 8 WMMA tiles),
// K stepped by 32. A[128][32] and B[32][128] staged row-major in LDS by
// async b128 copies, double buffered (stage k+1 while computing k).
// MODE 0: store f16            MODE 1: relu + store f16
// MODE 2: store f32            MODE 3: store f16 transposed as V^T[b][h][d][j]
// ---------------------------------------------------------------------------
template <int MODE>
__global__ __launch_bounds__(256)
void gemm_f16_wmma(const _Float16* __restrict__ A,
                   const _Float16* __restrict__ W,
                   const float* __restrict__ bias,
                   float* __restrict__ out32,
                   _Float16* __restrict__ out16,
                   int M, int K, int Nout) {
  __shared__ _Float16 Alds[2][128 * 32];   // 8 KB per buffer
  __shared__ _Float16 Blds[2][32 * 128];   // 8 KB per buffer (row-major [k][n])

  const int tid  = threadIdx.x;
  const int wave = tid >> 5;
  const int lane = tid & 31;
  const int ln15 = lane & 15;
  const int kg   = lane >> 4;              // 0 or 1 (half-wave group)

  const int m0 = blockIdx.y * 128;
  const int n0 = blockIdx.x * 128;

  // LDS byte offsets of the two buffers (generic ptr low 32 bits = LDS addr)
  const unsigned a_base[2] = {(unsigned)(uintptr_t)&Alds[0][0],
                              (unsigned)(uintptr_t)&Alds[1][0]};
  const unsigned b_base[2] = {(unsigned)(uintptr_t)&Blds[0][0],
                              (unsigned)(uintptr_t)&Blds[1][0]};

  // Stage one K-step tile pair into LDS buffer `buf` via async copies.
  auto stage = [&](int k0, int buf) {
    // A tile: 128 rows x 32 halves = 512 chunks of 8 halves (16 B)
#pragma unroll
    for (int c0 = 0; c0 < 2; ++c0) {
      const int c = tid + c0 * 256;
      const int ra = c >> 2, part = c & 3;
      async_copy_b128(a_base[buf] + (unsigned)(ra * 32 + part * 8) * 2,
                      A + (size_t)(m0 + ra) * K + k0 + part * 8);
    }
    // B tile: 32 rows x 128 halves = 512 chunks of 8 halves
#pragma unroll
    for (int c0 = 0; c0 < 2; ++c0) {
      const int c = tid + c0 * 256;
      const int rb = c >> 4, part = c & 15;
      async_copy_b128(b_base[buf] + (unsigned)(rb * 128 + part * 8) * 2,
                      W + (size_t)(k0 + rb) * Nout + n0 + part * 8);
    }
  };

  v8f acc[8];
#pragma unroll
  for (int nt = 0; nt < 8; ++nt) acc[nt] = (v8f)0.0f;

  stage(0, 0);
  wait_async_zero();
  __syncthreads();

  int buf = 0;
  for (int k0 = 0; k0 < K; k0 += 32, buf ^= 1) {
    if (k0 + 32 < K) stage(k0 + 32, buf ^ 1);   // overlap next fill w/ compute

    // ---- A fragment (16x32 f16): lane row M = ln15; K contiguous
    //      16*kg .. 16*kg+15 (matches B fragment) ----
    const _Float16* ap = &Alds[buf][(wave * 16 + ln15) * 32];
    h8 c0 = *(const h8*)(ap + 16 * kg);
    h8 c1 = *(const h8*)(ap + 16 * kg + 8);
    v16h afrag;
#pragma unroll
    for (int t = 0; t < 8; ++t) {
      afrag[t]     = c0[t];
      afrag[t + 8] = c1[t];
    }

#pragma unroll
    for (int nt = 0; nt < 8; ++nt) {
      // B fragment (32x16): lane col N = ln15, K rows kg*16 .. kg*16+15
      const _Float16* bp = &Blds[buf][kg * 16 * 128 + nt * 16 + ln15];
      v16h bfrag;
#pragma unroll
      for (int t = 0; t < 16; ++t) bfrag[t] = bp[t * 128];
      acc[nt] = __builtin_amdgcn_wmma_f32_16x16x32_f16(
          false, afrag, false, bfrag, (short)0, acc[nt], false, false);
    }

    wait_async_zero();   // next-tile async fills resident before anyone reads
    __syncthreads();
  }

  // ---- epilogue: C/D layout: M = kg*8 + r, N = ln15 ----
#pragma unroll
  for (int nt = 0; nt < 8; ++nt) {
#pragma unroll
    for (int r = 0; r < 8; ++r) {
      const int row = m0 + wave * 16 + kg * 8 + r;
      const int col = n0 + nt * 16 + ln15;
      float v = acc[nt][r] + bias[col];
      if (MODE == 0) {
        out16[(size_t)row * Nout + col] = (_Float16)v;
      } else if (MODE == 1) {
        out16[(size_t)row * Nout + col] = (_Float16)fmaxf(v, 0.0f);
      } else if (MODE == 2) {
        out32[(size_t)row * Nout + col] = v;
      } else {  // MODE 3: scatter to V^T[b][h][d][j]
        const int b = row >> 9, j = row & 511;
        const int h = col >> 6, d = col & 63;
        out16[(((size_t)(b * HEADS + h) * HEAD_DIM + d) * NSEQ) + j] =
            (_Float16)v;
      }
    }
  }
}

// ---------------------------------------------------------------------------
// Attention: one wave per (b, head, 16-row query tile).
// Phase 1: S = Q K^T via WMMA (16x512 f32 tile in LDS)
// Phase 2: scale + spatial/edge bias + key mask + softmax (shuffle reduce)
// Phase 3: O = P V via WMMA (P converted f32->f16 from LDS on the fly)
// ---------------------------------------------------------------------------
__global__ __launch_bounds__(32)
void attn_wmma(const _Float16* __restrict__ qh,
               const _Float16* __restrict__ kh,
               const _Float16* __restrict__ vT,
               const int* __restrict__ dist,
               const float* __restrict__ edge,
               const unsigned char* __restrict__ mask,
               const float* __restrict__ spatial,
               const float* __restrict__ edge_w,
               const float* __restrict__ edge_b,
               _Float16* __restrict__ ctx) {
  __shared__ float S[16][NSEQ];   // 32 KB score tile
  __shared__ float rsum[16];

  const int it   = blockIdx.x;    // query tile (0..31)
  const int head = blockIdx.y;
  const int b    = blockIdx.z;
  const int i0   = it * 16;
  const int lane = threadIdx.x;
  const int ln15 = lane & 15;
  const int kg   = lane >> 4;

  // ---- Q fragments (row = i0 + ln15, K-dim = head dim, 2 steps of 32) ----
  const _Float16* qrow =
      qh + ((size_t)(b * NSEQ + i0 + ln15) * HIDDEN) + head * HEAD_DIM;
  v16h aq[2];
#pragma unroll
  for (int kk = 0; kk < 2; ++kk) {
    const int koff = kk * 32;
    h8 c0 = *(const h8*)(qrow + koff + 16 * kg);
    h8 c1 = *(const h8*)(qrow + koff + 16 * kg + 8);
#pragma unroll
    for (int t = 0; t < 8; ++t) {
      aq[kk][t]     = c0[t];
      aq[kk][t + 8] = c1[t];
    }
  }

  // ---- Phase 1: scores ----
  for (int jt = 0; jt < NSEQ / 16; ++jt) {
    const _Float16* krow =
        kh + ((size_t)(b * NSEQ + jt * 16 + ln15) * HIDDEN) + head * HEAD_DIM +
        kg * 16;
    v16h bk0 = *(const v16h*)(krow);        // K-dim 0..31
    v16h bk1 = *(const v16h*)(krow + 32);   // K-dim 32..63
    v8f acc = (v8f)0.0f;
    acc = __builtin_amdgcn_wmma_f32_16x16x32_f16(false, aq[0], false, bk0,
                                                 (short)0, acc, false, false);
    acc = __builtin_amdgcn_wmma_f32_16x16x32_f16(false, aq[1], false, bk1,
                                                 (short)0, acc, false, false);
#pragma unroll
    for (int r = 0; r < 8; ++r) S[kg * 8 + r][jt * 16 + ln15] = acc[r];
  }
  __syncthreads();

  // ---- Phase 2: bias + mask ----
  const float ew = edge_w[head], eb = edge_b[head];
  for (int idx = lane; idx < 16 * NSEQ; idx += 32) {
    const int row = idx >> 9, j = idx & 511;
    const size_t ij = ((size_t)b * NSEQ + (i0 + row)) * NSEQ + j;
    float s = S[row][j] * 0.125f;           // 1/sqrt(64)
    s += spatial[dist[ij] * HEADS + head] + edge[ij] * ew + eb;
    if (mask[b * NSEQ + j]) s = -__builtin_inff();
    S[row][j] = s;
  }
  __syncthreads();

  // ---- softmax (unnormalized exp stored; rowsum kept for final scale) ----
  for (int row = 0; row < 16; ++row) {
    float mx = -__builtin_inff();
    for (int j = lane; j < NSEQ; j += 32) mx = fmaxf(mx, S[row][j]);
#pragma unroll
    for (int off = 16; off > 0; off >>= 1)
      mx = fmaxf(mx, __shfl_xor(mx, off, 32));
    float sm = 0.0f;
    for (int j = lane; j < NSEQ; j += 32) {
      float p = __expf(S[row][j] - mx);
      S[row][j] = p;
      sm += p;
    }
#pragma unroll
    for (int off = 16; off > 0; off >>= 1) sm += __shfl_xor(sm, off, 32);
    if (lane == 0) rsum[row] = sm;
  }
  __syncthreads();

  // ---- Phase 3: O = P @ V ----
  v8f oacc[4];
#pragma unroll
  for (int dt = 0; dt < 4; ++dt) oacc[dt] = (v8f)0.0f;

  for (int ks = 0; ks < NSEQ / 32; ++ks) {
    const int j0 = ks * 32;
    v16h ap;
#pragma unroll
    for (int t = 0; t < 16; ++t) {
      const int kidx = t + 16 * kg;
      ap[t] = (_Float16)S[ln15][j0 + kidx];
    }
#pragma unroll
    for (int dt = 0; dt < 4; ++dt) {
      const _Float16* vp =
          vT + (((size_t)(b * HEADS + head) * HEAD_DIM + dt * 16 + ln15) *
                NSEQ) + j0 + kg * 16;
      v16h bv = *(const v16h*)vp;
      oacc[dt] = __builtin_amdgcn_wmma_f32_16x16x32_f16(
          false, ap, false, bv, (short)0, oacc[dt], false, false);
    }
  }

  // ---- store ctx[b, i, h*64 + d], normalized by rowsum ----
#pragma unroll
  for (int dt = 0; dt < 4; ++dt) {
#pragma unroll
    for (int r = 0; r < 8; ++r) {
      const int row = kg * 8 + r;
      const float inv = 1.0f / rsum[row];
      ctx[((size_t)(b * NSEQ + i0 + row)) * HIDDEN + head * HEAD_DIM +
          dt * 16 + ln15] = (_Float16)(oacc[dt][r] * inv);
    }
  }
}

// ---------------------------------------------------------------------------
// y = LayerNorm(a + resid) * g + beta ; one block per row of 512
// ---------------------------------------------------------------------------
__global__ __launch_bounds__(256)
void addln_kernel(const float* __restrict__ a, const float* __restrict__ resid,
                  const float* __restrict__ g, const float* __restrict__ bta,
                  float* __restrict__ y32, _Float16* __restrict__ y16) {
  __shared__ float red[256];
  const int row = blockIdx.x, tid = threadIdx.x;
  const size_t base = (size_t)row * HIDDEN;

  float v0 = a[base + tid] + resid[base + tid];
  float v1 = a[base + tid + 256] + resid[base + tid + 256];

  red[tid] = v0 + v1;
  __syncthreads();
  for (int st = 128; st > 0; st >>= 1) {
    if (tid < st) red[tid] += red[tid + st];
    __syncthreads();
  }
  const float mu = red[0] * (1.0f / HIDDEN);
  __syncthreads();

  const float d0 = v0 - mu, d1 = v1 - mu;
  red[tid] = d0 * d0 + d1 * d1;
  __syncthreads();
  for (int st = 128; st > 0; st >>= 1) {
    if (tid < st) red[tid] += red[tid + st];
    __syncthreads();
  }
  const float rstd = rsqrtf(red[0] * (1.0f / HIDDEN) + 1e-5f);

  const float y0 = d0 * rstd * g[tid] + bta[tid];
  const float y1 = d1 * rstd * g[tid + 256] + bta[tid + 256];
  y32[base + tid] = y0;
  y32[base + tid + 256] = y1;
  if (y16) {
    y16[base + tid] = (_Float16)y0;
    y16[base + tid + 256] = (_Float16)y1;
  }
}

// ---------------------------------------------------------------------------
// Host launcher
// ---------------------------------------------------------------------------
extern "C" void kernel_launch(void* const* d_in, const int* in_sizes, int n_in,
                              void* d_out, int out_size, void* d_ws,
                              size_t ws_size, hipStream_t stream) {
  const float* x        = (const float*)d_in[0];
  const int*   dist     = (const int*)d_in[1];
  const float* edge     = (const float*)d_in[2];
  const unsigned char* mask = (const unsigned char*)d_in[3];  // jax bool = u8
  const float* Wq = (const float*)d_in[4];  const float* bq = (const float*)d_in[5];
  const float* Wk = (const float*)d_in[6];  const float* bk = (const float*)d_in[7];
  const float* Wv = (const float*)d_in[8];  const float* bv = (const float*)d_in[9];
  const float* Wo = (const float*)d_in[10]; const float* bo = (const float*)d_in[11];
  const float* spatial = (const float*)d_in[12];
  const float* edge_w  = (const float*)d_in[13];
  const float* edge_b  = (const float*)d_in[14];
  const float* ln1_g = (const float*)d_in[15]; const float* ln1_b = (const float*)d_in[16];
  const float* ln2_g = (const float*)d_in[17]; const float* ln2_b = (const float*)d_in[18];
  const float* W1 = (const float*)d_in[19]; const float* b1 = (const float*)d_in[20];
  const float* W2 = (const float*)d_in[21]; const float* b2 = (const float*)d_in[22];

  char* ws = (char*)d_ws;
  const size_t S_ACT16 = (size_t)MROWS * HIDDEN * 2;        // 8 MiB
  const size_t S_W512  = (size_t)HIDDEN * HIDDEN * 2;       // 512 KiB
  const size_t S_WF    = (size_t)HIDDEN * FFN_DIM * 2;      // 2 MiB
  const size_t S_ACT32 = (size_t)MROWS * HIDDEN * 4;        // 16 MiB

  size_t off = 0;
  _Float16* xh  = (_Float16*)(ws + off); off += S_ACT16;
  _Float16* Wqh = (_Float16*)(ws + off); off += S_W512;
  _Float16* Wkh = (_Float16*)(ws + off); off += S_W512;
  _Float16* Wvh = (_Float16*)(ws + off); off += S_W512;
  _Float16* Woh = (_Float16*)(ws + off); off += S_W512;
  _Float16* W1h = (_Float16*)(ws + off); off += S_WF;
  _Float16* W2h = (_Float16*)(ws + off); off += S_WF;
  size_t qh_off = off;
  _Float16* qh  = (_Float16*)(ws + off); off += S_ACT16;
  _Float16* kh  = (_Float16*)(ws + off); off += S_ACT16;
  _Float16* vT  = (_Float16*)(ws + off); off += S_ACT16;
  _Float16* ctx = (_Float16*)(ws + off); off += S_ACT16;
  size_t o32_off = off;
  float*    o32 = (float*)(ws + off); off += S_ACT32;
  float*    h32 = (float*)(ws + off); off += S_ACT32;
  _Float16* hh  = (_Float16*)(ws + off); off += S_ACT16;
  // Aliased (lifetimes disjoint):
  _Float16* u16   = (_Float16*)(ws + qh_off);   // 32 MiB over qh/kh/vT/ctx
  float*    f2out = (float*)(ws + o32_off);     // 16 MiB over o32

  // ---- f32 -> f16 conversions ----
  cvt_f32_f16<<<2048, 256, 0, stream>>>(x, xh, MROWS * HIDDEN);
  cvt_f32_f16<<<1024, 256, 0, stream>>>(Wq, Wqh, HIDDEN * HIDDEN);
  cvt_f32_f16<<<1024, 256, 0, stream>>>(Wk, Wkh, HIDDEN * HIDDEN);
  cvt_f32_f16<<<1024, 256, 0, stream>>>(Wv, Wvh, HIDDEN * HIDDEN);
  cvt_f32_f16<<<1024, 256, 0, stream>>>(Wo, Woh, HIDDEN * HIDDEN);
  cvt_f32_f16<<<2048, 256, 0, stream>>>(W1, W1h, HIDDEN * FFN_DIM);
  cvt_f32_f16<<<2048, 256, 0, stream>>>(W2, W2h, FFN_DIM * HIDDEN);

  const dim3 blk(256);
  const dim3 g512(HIDDEN / 128, MROWS / 128);   // (4, 64)
  const dim3 gffn(FFN_DIM / 128, MROWS / 128);  // (16, 64)

  // ---- QKV projections ----
  gemm_f16_wmma<0><<<g512, blk, 0, stream>>>(xh, Wqh, bq, nullptr, qh,
                                             MROWS, HIDDEN, HIDDEN);
  gemm_f16_wmma<0><<<g512, blk, 0, stream>>>(xh, Wkh, bk, nullptr, kh,
                                             MROWS, HIDDEN, HIDDEN);
  gemm_f16_wmma<3><<<g512, blk, 0, stream>>>(xh, Wvh, bv, nullptr, vT,
                                             MROWS, HIDDEN, HIDDEN);

  // ---- attention ----
  attn_wmma<<<dim3(NSEQ / 16, HEADS, NB), dim3(32), 0, stream>>>(
      qh, kh, vT, dist, edge, mask, spatial, edge_w, edge_b, ctx);

  // ---- output projection + LN1 ----
  gemm_f16_wmma<2><<<g512, blk, 0, stream>>>(ctx, Woh, bo, o32, nullptr,
                                             MROWS, HIDDEN, HIDDEN);
  addln_kernel<<<MROWS, 256, 0, stream>>>(o32, x, ln1_g, ln1_b, h32, hh);

  // ---- FFN + LN2 ----
  gemm_f16_wmma<1><<<gffn, blk, 0, stream>>>(hh, W1h, b1, nullptr, u16,
                                             MROWS, HIDDEN, FFN_DIM);
  gemm_f16_wmma<2><<<g512, blk, 0, stream>>>(u16, W2h, b2, f2out, nullptr,
                                             MROWS, FFN_DIM, HIDDEN);
  addln_kernel<<<MROWS, 256, 0, stream>>>(f2out, h32, ln2_g, ln2_b,
                                          (float*)d_out, nullptr);

  (void)in_sizes; (void)n_in; (void)out_size; (void)ws_size;
}
